// CQAttention_3693671875124
// MI455X (gfx1250) — matrix operand order repaired and checked by reading, not compiled, not run on hardware
//
#include <hip/hip_runtime.h>
#include <hip/hip_bf16.h>
#include <cstdint>

typedef __attribute__((ext_vector_type(16))) _Float16 v16h;
typedef __attribute__((ext_vector_type(8)))  float    v8f;

#define B_   64
#define LC   2048
#define LQ   256
#define DD   128
#define NEGV (-1e30f)

__device__ __forceinline__ v8f wmma_f16(v16h a, v16h b, v8f c) {
  // D = A(16x32 f16) * B(32x16 f16) + C(16x16 f32)
  return __builtin_amdgcn_wmma_f32_16x16x32_f16(false, a, false, b, (short)0, c, false, false);
}

// ---- CDNA5 async LDS staging (ASYNCcnt path) ----
typedef __attribute__((address_space(3))) char lds_char_t;
__device__ __forceinline__ uint32_t lds_off32(const void* p) {
  return (uint32_t)(uintptr_t)(lds_char_t*)(p);
}
__device__ __forceinline__ void async_copy_b128(uint32_t lds_byte_addr, const void* gaddr) {
  asm volatile("global_load_async_to_lds_b128 %0, %1, off"
               :: "v"(lds_byte_addr), "v"(gaddr) : "memory");
}
__device__ __forceinline__ void wait_async0() {
  asm volatile("s_wait_asynccnt 0" ::: "memory");
}

// A operand: M=16 rows, K=32.  element (m,k) at p[m*mS + k*kS]
// CDNA5 16-bit A layout: lane half h, VGPR v, pair p: K = (v<4 ? 8h+2v : 16+8h+2(v-4)) + p
__device__ __forceinline__ v16h ld_a(const _Float16* p, int mS, int kS) {
  const int lane = threadIdx.x & 31;
  const int half = lane >> 4;
  const int row  = lane & 15;
  v16h a;
#pragma unroll
  for (int j = 0; j < 16; ++j) {
    const int v = j >> 1, pr = j & 1;
    const int k = (v < 4) ? (8*half + 2*v + pr) : (16 + 8*half + 2*(v-4) + pr);
    a[j] = p[row*mS + k*kS];
  }
  return a;
}

// B operand: K=32 rows, N=16 cols.  element (k,n) at p[k*kS + n*nS]
// CDNA5 16-bit B layout: element j of lane half h holds K = (j<8 ? 8h+j : 16+8h+(j-8))
__device__ __forceinline__ v16h ld_b(const _Float16* p, int kS, int nS) {
  const int lane = threadIdx.x & 31;
  const int half = lane >> 4;
  const int col  = lane & 15;
  v16h b;
#pragma unroll
  for (int j = 0; j < 16; ++j) {
    const int k = (j < 8) ? (8*half + j) : (16 + 8*half + (j - 8));
    b[j] = p[k*kS + col*nS];
  }
  return b;
}

// ---------------- K0: per-row linear terms + masks folded; Q -> f16 ----------------
__global__ __launch_bounds__(256) void k0_prep(
    const float* __restrict__ C, const float* __restrict__ Q,
    const float* __restrict__ Cmask, const float* __restrict__ Qmask,
    const float* __restrict__ w4C, const float* __restrict__ w4Q,
    const float* __restrict__ bias,
    float* __restrict__ cv, float* __restrict__ cvn,
    float* __restrict__ qvb, float* __restrict__ qvn,
    _Float16* __restrict__ Qh_g) {
  const int b = blockIdx.x;
  const int tid = threadIdx.x;
  __shared__ float wc[DD], wq[DD];
  if (tid < DD) { wc[tid] = w4C[tid]; wq[tid] = w4Q[tid]; }
  __syncthreads();
  const float b0 = bias[0];
  for (int c = tid; c < LC; c += 256) {
    const float* row = C + ((size_t)b*LC + c)*DD;
    float s = 0.f;
#pragma unroll 8
    for (int d = 0; d < DD; ++d) s += row[d]*wc[d];
    cv[b*LC + c]  = s;
    cvn[b*LC + c] = s + NEGV*(1.0f - Cmask[b*LC + c]);
  }
  for (int q = tid; q < LQ; q += 256) {
    const float* row = Q + ((size_t)b*LQ + q)*DD;
    float s = 0.f;
#pragma unroll 8
    for (int d = 0; d < DD; ++d) s += row[d]*wq[d];
    qvb[b*LQ + q] = s + b0;
    qvn[b*LQ + q] = s + b0 + NEGV*(1.0f - Qmask[b*LQ + q]);
  }
  for (int i = tid; i < LQ*DD; i += 256)
    Qh_g[(size_t)b*LQ*DD + i] = (_Float16)Q[(size_t)b*LQ*DD + i];
}

// ---------------- K1: online column-softmax stats (m2, z2 per q) ----------------
__global__ __launch_bounds__(256) void k1_colstats(
    const float* __restrict__ C, const _Float16* __restrict__ Qh_g,
    const float* __restrict__ w4mlu,
    const float* __restrict__ cvn, const float* __restrict__ qvb,
    float* __restrict__ m2, float* __restrict__ z2) {
  extern __shared__ char smem[];
  _Float16* Qh  = (_Float16*)smem;        // LQ*DD
  _Float16* Cwh = Qh + LQ*DD;             // 16*DD
  const int b = blockIdx.x;
  const int tid = threadIdx.x;
  const int lane = tid & 31;
  const int wave = tid >> 5;
  // async-stage Qh (pure f16 copy, ASYNCcnt path, no VGPR round trip)
  {
    const char* src = (const char*)(Qh_g + (size_t)b*LQ*DD);
    const uint32_t dst = lds_off32(Qh);
    for (int i = tid*16; i < LQ*DD*2; i += 256*16)
      async_copy_b128(dst + (uint32_t)i, src + i);
    wait_async0();
  }
  const int q0 = wave*32;   // each wave owns 32 q columns
  const float qv0 = qvb[b*LQ + q0 + (lane & 15)];
  const float qv1 = qvb[b*LQ + q0 + 16 + (lane & 15)];
  float m0 = -3.0e38f, z0 = 0.f, m1 = -3.0e38f, z1 = 0.f;
  for (int c0 = 0; c0 < LC; c0 += 16) {
    __syncthreads();
    for (int i = tid; i < 16*DD; i += 256) {
      const int d = i & (DD-1);
      Cwh[i] = (_Float16)(C[((size_t)b*LC + c0)*DD + i] * w4mlu[d]);
    }
    if (c0 + 16 < LC && tid < 64)  // prefetch next c-tile (64 x 128B lines)
      __builtin_prefetch(C + ((size_t)b*LC + c0 + 16)*DD + tid*32, 0, 3);
    __syncthreads();
    v8f a0 = {}; v8f a1 = {};
#pragma unroll
    for (int kk = 0; kk < DD; kk += 32) {
      v16h av  = ld_a(Cwh + kk, DD, 1);
      v16h b0v = ld_b(Qh + (size_t)q0*DD + kk, 1, DD);
      v16h b1v = ld_b(Qh + (size_t)(q0+16)*DD + kk, 1, DD);
      a0 = wmma_f16(av, b0v, a0);
      a1 = wmma_f16(av, b1v, a1);
    }
    float v0[8], v1[8];
#pragma unroll
    for (int r = 0; r < 8; ++r) {
      const int c = c0 + r + 8*(lane >> 4);
      const float cc = cvn[b*LC + c];
      v0[r] = a0[r] + cc + qv0;
      v1[r] = a1[r] + cc + qv1;
    }
    float mt0 = v0[0], mt1 = v1[0];
#pragma unroll
    for (int r = 1; r < 8; ++r) { mt0 = fmaxf(mt0, v0[r]); mt1 = fmaxf(mt1, v1[r]); }
    mt0 = fmaxf(mt0, __shfl_xor(mt0, 16, 32));
    mt1 = fmaxf(mt1, __shfl_xor(mt1, 16, 32));
    float zt0 = 0.f, zt1 = 0.f;
#pragma unroll
    for (int r = 0; r < 8; ++r) { zt0 += __expf(v0[r]-mt0); zt1 += __expf(v1[r]-mt1); }
    zt0 += __shfl_xor(zt0, 16, 32);
    zt1 += __shfl_xor(zt1, 16, 32);
    const float nm0 = fmaxf(m0, mt0);
    z0 = z0*__expf(m0-nm0) + zt0*__expf(mt0-nm0); m0 = nm0;
    const float nm1 = fmaxf(m1, mt1);
    z1 = z1*__expf(m1-nm1) + zt1*__expf(mt1-nm1); m1 = nm1;
  }
  const int qc = q0 + 16*(lane>>4) + (lane&15);
  m2[b*LQ + qc] = (lane < 16) ? m0 : m1;
  z2[b*LQ + qc] = (lane < 16) ? z0 : z1;
}

// ---------------- K2: T = S2^T @ C  (256x128 per batch, stored f16) ----------------
__global__ __launch_bounds__(256) void k2_T(
    const float* __restrict__ C, const _Float16* __restrict__ Qh_g,
    const float* __restrict__ w4mlu,
    const float* __restrict__ cvn, const float* __restrict__ qvb,
    const float* __restrict__ m2, const float* __restrict__ z2,
    _Float16* __restrict__ Th_g) {
  extern __shared__ char smem[];
  _Float16* Qh  = (_Float16*)smem;        // LQ*DD
  _Float16* Cwh = Qh + LQ*DD;             // 32*DD
  _Float16* Ch  = Cwh + 32*DD;            // 32*DD
  _Float16* p2t = Ch + 32*DD;             // [q:LQ][c:32]
  const int b = blockIdx.x;
  const int tid = threadIdx.x;
  const int lane = tid & 31;
  const int wave = tid >> 5;
  {
    const char* src = (const char*)(Qh_g + (size_t)b*LQ*DD);
    const uint32_t dst = lds_off32(Qh);
    for (int i = tid*16; i < LQ*DD*2; i += 256*16)
      async_copy_b128(dst + (uint32_t)i, src + i);
    wait_async0();
  }
  const int q0 = wave*32;
  const int qa  = q0 + (lane & 15);
  const int qb_ = q0 + 16 + (lane & 15);
  const float qv0 = qvb[b*LQ + qa], qv1 = qvb[b*LQ + qb_];
  const float m20 = m2[b*LQ + qa],  m21 = m2[b*LQ + qb_];
  const float iz0 = 1.f / z2[b*LQ + qa], iz1 = 1.f / z2[b*LQ + qb_];
  v8f zz = {};
  v8f t[2][8];
#pragma unroll
  for (int i = 0; i < 2; ++i)
#pragma unroll
    for (int j = 0; j < 8; ++j) t[i][j] = zz;
  for (int c0 = 0; c0 < LC; c0 += 32) {
    __syncthreads();
    for (int i = tid; i < 32*DD; i += 256) {
      const int d = i & (DD-1);
      const float cf = C[((size_t)b*LC + c0)*DD + i];
      Ch[i]  = (_Float16)cf;
      Cwh[i] = (_Float16)(cf * w4mlu[d]);
    }
    if (c0 + 32 < LC && tid < 128)  // prefetch next c-tile (128 x 128B lines)
      __builtin_prefetch(C + ((size_t)b*LC + c0 + 32)*DD + tid*32, 0, 3);
    __syncthreads();
    v8f s00 = zz, s01 = zz, s10 = zz, s11 = zz;
#pragma unroll
    for (int kk = 0; kk < DD; kk += 32) {
      v16h a0 = ld_a(Cwh + kk, DD, 1);
      v16h a1 = ld_a(Cwh + 16*DD + kk, DD, 1);
      v16h b0 = ld_b(Qh + (size_t)q0*DD + kk, 1, DD);
      v16h b1 = ld_b(Qh + (size_t)(q0+16)*DD + kk, 1, DD);
      s00 = wmma_f16(a0, b0, s00);
      s01 = wmma_f16(a0, b1, s01);
      s10 = wmma_f16(a1, b0, s10);
      s11 = wmma_f16(a1, b1, s11);
    }
#pragma unroll
    for (int r = 0; r < 8; ++r) {
      const int cA = r + 8*(lane>>4);
      const int cB = 16 + cA;
      const float ccA = cvn[b*LC + c0 + cA];
      const float ccB = cvn[b*LC + c0 + cB];
      p2t[qa*32  + cA] = (_Float16)(__expf(s00[r] + ccA + qv0 - m20) * iz0);
      p2t[qb_*32 + cA] = (_Float16)(__expf(s01[r] + ccA + qv1 - m21) * iz1);
      p2t[qa*32  + cB] = (_Float16)(__expf(s10[r] + ccB + qv0 - m20) * iz0);
      p2t[qb_*32 + cB] = (_Float16)(__expf(s11[r] + ccB + qv1 - m21) * iz1);
    }
    __syncthreads();
    // accumulate T (q-strip x all d) : A = P2^T tile from LDS, B = C tile
    v16h aT0 = ld_a(p2t + q0*32, 32, 1);
    v16h aT1 = ld_a(p2t + (q0+16)*32, 32, 1);
#pragma unroll
    for (int nt = 0; nt < 8; ++nt) {
      v16h bC = ld_b(Ch + nt*16, DD, 1);
      t[0][nt] = wmma_f16(aT0, bC, t[0][nt]);
      t[1][nt] = wmma_f16(aT1, bC, t[1][nt]);
    }
  }
#pragma unroll
  for (int mt = 0; mt < 2; ++mt)
#pragma unroll
    for (int nt = 0; nt < 8; ++nt)
#pragma unroll
      for (int r = 0; r < 8; ++r) {
        const int q = q0 + mt*16 + r + 8*(lane>>4);
        const int d = nt*16 + (lane & 15);
        Th_g[((size_t)b*LQ + q)*DD + d] = (_Float16)t[mt][nt][r];
      }
}

// ---------------- K3: fused S -> row softmax -> A, Bt -> concat/transpose out ----------------
__global__ __launch_bounds__(256) void k3_main(
    const float* __restrict__ C, const _Float16* __restrict__ Qh_g,
    const float* __restrict__ w4mlu,
    const float* __restrict__ cv, const float* __restrict__ qvn,
    const _Float16* __restrict__ Th_g,
    float* __restrict__ out) {
  extern __shared__ char smem[];
  _Float16* Qh  = (_Float16*)smem;        // LQ*DD
  _Float16* Th  = Qh + LQ*DD;             // LQ*DD
  _Float16* Cwh = Th + LQ*DD;             // 64*DD
  _Float16* p1  = Cwh + 64*DD;            // [c:64][q:LQ]
  float* Ct   = (float*)(p1 + 64*LQ);     // 64*DD
  float* As   = Ct + 64*DD;               // 64*DD
  float* Bs   = As + 64*DD;               // 64*DD
  float* rowm = Bs + 64*DD;               // [2][64]
  float* rowz = rowm + 128;               // [2][64]
  const int ct = blockIdx.x;              // c-tile (64 rows)
  const int b  = blockIdx.y;
  const int c_base = ct*64;
  const int tid = threadIdx.x;
  const int lane = tid & 31;
  const int wave = tid >> 5;
  const int mtile = wave >> 1;            // 0..3 (16 c rows each)
  const int qhalf = wave & 1;             // 0..1 (128 q each)

  // async-stage Qh and Th (both already f16 in global)
  {
    const char* qsrc = (const char*)(Qh_g + (size_t)b*LQ*DD);
    const char* tsrc = (const char*)(Th_g + (size_t)b*LQ*DD);
    const uint32_t qdst = lds_off32(Qh);
    const uint32_t tdst = lds_off32(Th);
    for (int i = tid*16; i < LQ*DD*2; i += 256*16) {
      async_copy_b128(qdst + (uint32_t)i, qsrc + i);
      async_copy_b128(tdst + (uint32_t)i, tsrc + i);
    }
  }
  for (int i = tid; i < 64*DD; i += 256) {
    const int d = i & (DD-1);
    const float cf = C[((size_t)b*LC + c_base)*DD + i];
    Ct[i]  = cf;
    Cwh[i] = (_Float16)(cf * w4mlu[d]);
  }
  wait_async0();
  __syncthreads();

  // S tiles: this wave = rows [mtile*16,+16) x q [qhalf*128,+128)
  v8f zz = {};
  v8f acc[8];
#pragma unroll
  for (int nt = 0; nt < 8; ++nt) acc[nt] = zz;
#pragma unroll
  for (int kk = 0; kk < DD; kk += 32) {
    v16h av = ld_a(Cwh + mtile*16*DD + kk, DD, 1);
#pragma unroll
    for (int nt = 0; nt < 8; ++nt) {
      v16h bv = ld_b(Qh + (size_t)(qhalf*128 + nt*16)*DD + kk, 1, DD);
      acc[nt] = wmma_f16(av, bv, acc[nt]);
    }
  }
  float qv[8];
#pragma unroll
  for (int nt = 0; nt < 8; ++nt) qv[nt] = qvn[b*LQ + qhalf*128 + nt*16 + (lane & 15)];
  float cvr[8];
#pragma unroll
  for (int r = 0; r < 8; ++r) cvr[r] = cv[b*LC + c_base + mtile*16 + r + 8*(lane>>4)];
#pragma unroll
  for (int nt = 0; nt < 8; ++nt)
#pragma unroll
    for (int r = 0; r < 8; ++r) acc[nt][r] += cvr[r] + qv[nt];

  // partial row softmax over this wave's 128 q (reduce over ntiles + lanes within 16-group)
  float lm[8], lz[8];
#pragma unroll
  for (int r = 0; r < 8; ++r) {
    float m = acc[0][r];
#pragma unroll
    for (int nt = 1; nt < 8; ++nt) m = fmaxf(m, acc[nt][r]);
    m = fmaxf(m, __shfl_xor(m, 1, 32));
    m = fmaxf(m, __shfl_xor(m, 2, 32));
    m = fmaxf(m, __shfl_xor(m, 4, 32));
    m = fmaxf(m, __shfl_xor(m, 8, 32));
    lm[r] = m;
    float z = 0.f;
#pragma unroll
    for (int nt = 0; nt < 8; ++nt) z += __expf(acc[nt][r] - m);
    z += __shfl_xor(z, 1, 32);
    z += __shfl_xor(z, 2, 32);
    z += __shfl_xor(z, 4, 32);
    z += __shfl_xor(z, 8, 32);
    lz[r] = z;
  }
  if ((lane & 15) == 0) {
#pragma unroll
    for (int r = 0; r < 8; ++r) {
      const int rr = mtile*16 + r + 8*(lane>>4);
      rowm[qhalf*64 + rr] = lm[r];
      rowz[qhalf*64 + rr] = lz[r];
    }
  }
  __syncthreads();
  // merge the two q-halves, write normalized P1 (f16) to LDS
#pragma unroll
  for (int r = 0; r < 8; ++r) {
    const int rr = mtile*16 + r + 8*(lane>>4);
    const float ma = rowm[rr], mb = rowm[64 + rr];
    const float za = rowz[rr], zb = rowz[64 + rr];
    const float M = fmaxf(ma, mb);
    const float Z = za*__expf(ma - M) + zb*__expf(mb - M);
    const float invZ = 1.f / Z;
#pragma unroll
    for (int nt = 0; nt < 8; ++nt) {
      const int q = qhalf*128 + nt*16 + (lane & 15);
      p1[rr*LQ + q] = (_Float16)(__expf(acc[nt][r] - M) * invZ);
    }
  }
  __syncthreads();

  // A = P1 @ Q ; Bt = P1 @ T  — wave = rows mtile x d-half (4 ntiles), shared A operand
  const int nhalf = qhalf;
  v8f aA[4], aB[4];
#pragma unroll
  for (int j = 0; j < 4; ++j) { aA[j] = zz; aB[j] = zz; }
#pragma unroll
  for (int kq = 0; kq < LQ; kq += 32) {
    v16h ap = ld_a(p1 + mtile*16*LQ + kq, LQ, 1);
#pragma unroll
    for (int j = 0; j < 4; ++j) {
      const int d0 = nhalf*64 + j*16;
      v16h bq = ld_b(Qh + (size_t)kq*DD + d0, DD, 1);
      v16h bt = ld_b(Th + (size_t)kq*DD + d0, DD, 1);
      aA[j] = wmma_f16(ap, bq, aA[j]);
      aB[j] = wmma_f16(ap, bt, aB[j]);
    }
  }
#pragma unroll
  for (int j = 0; j < 4; ++j)
#pragma unroll
    for (int r = 0; r < 8; ++r) {
      const int cc = mtile*16 + r + 8*(lane>>4);
      const int d  = nhalf*64 + j*16 + (lane & 15);
      As[cc*DD + d] = aA[j][r];
      Bs[cc*DD + d] = aB[j][r];
    }
  __syncthreads();

  // out[b, f, c]: f in [0,512): [C ; A ; C*A ; C*Bt], coalesced along c
  const int cl = tid & 63;
  const int fg = tid >> 6;
  for (int f = fg; f < 512; f += 4) {
    const int sec = f >> 7;
    const int d = f & (DD-1);
    const float cvx = Ct[cl*DD + d];
    float v;
    if (sec == 0)      v = cvx;
    else if (sec == 1) v = As[cl*DD + d];
    else if (sec == 2) v = cvx * As[cl*DD + d];
    else               v = cvx * Bs[cl*DD + d];
    out[((size_t)b*512 + f)*LC + c_base + cl] = v;
  }
}

extern "C" void kernel_launch(void* const* d_in, const int* in_sizes, int n_in,
                              void* d_out, int out_size, void* d_ws, size_t ws_size,
                              hipStream_t stream) {
  const float* C     = (const float*)d_in[0];
  const float* Q     = (const float*)d_in[1];
  const float* Cmask = (const float*)d_in[2];
  const float* Qmask = (const float*)d_in[3];
  const float* w4C   = (const float*)d_in[4];
  const float* w4Q   = (const float*)d_in[5];
  const float* w4mlu = (const float*)d_in[6];
  const float* bias  = (const float*)d_in[7];
  float* out = (float*)d_out;

  // workspace carve (~10 MB)
  float* cv  = (float*)d_ws;
  float* cvn = cv  + B_*LC;
  float* qvb = cvn + B_*LC;
  float* qvn = qvb + B_*LQ;
  float* m2  = qvn + B_*LQ;
  float* z2  = m2  + B_*LQ;
  _Float16* Th  = (_Float16*)(z2 + B_*LQ);   // B*LQ*DD f16
  _Float16* Qhg = Th + (size_t)B_*LQ*DD;     // B*LQ*DD f16

  k0_prep<<<dim3(B_), dim3(256), 0, stream>>>(C, Q, Cmask, Qmask, w4C, w4Q, bias,
                                              cv, cvn, qvb, qvn, Qhg);
  const size_t s1 = (size_t)(LQ*DD + 16*DD) * sizeof(_Float16);
  k1_colstats<<<dim3(B_), dim3(256), s1, stream>>>(C, Qhg, w4mlu, cvn, qvb, m2, z2);
  const size_t s2 = (size_t)(LQ*DD + 32*DD + 32*DD + LQ*32) * sizeof(_Float16);
  k2_T<<<dim3(B_), dim3(256), s2, stream>>>(C, Qhg, w4mlu, cvn, qvb, m2, z2, Th);
  const size_t s3 = (size_t)(LQ*DD + LQ*DD + 64*DD + 64*LQ) * sizeof(_Float16)
                  + (size_t)(64*DD*3 + 256) * sizeof(float);
  k3_main<<<dim3(32, B_), dim3(256), s3, stream>>>(C, Qhg, w4mlu, cv, qvn, Th, out);
}
